// Net_57277683859526
// MI455X (gfx1250) — matrix operand order repaired and checked by reading, not compiled, not run on hardware
//
#include <hip/hip_runtime.h>
#include <hip/hip_bf16.h>

// CDNA5 / gfx1250: wave32, v_wmma_f32_16x16x32_f16.
typedef _Float16 v16h __attribute__((ext_vector_type(16)));
typedef _Float16 h8   __attribute__((ext_vector_type(8)));
typedef float    v8f  __attribute__((ext_vector_type(8)));

#define DIMQ     16
#define EEXP     256           // DIM*DIM experts
#define D_IN     15
#define HID      128
#define B_TOT    8192
#define BT       128           // batch rows per workgroup
#define NTILES   (B_TOT / BT)  // 64
#define KPAD     32            // D_IN padded to one WMMA K-step

// LDS strides in halfs (padded for bank-conflict-free fragment access)
#define SA_STR   40            // x rows     [BT][40]   (batch-major, K-padded)
#define SW1_STR  40            // W1^T rows  [HID][40]  (out-major, k contiguous)
#define SW2_STR  136           // W2^T rows  [HID][136]
#define SH_STR   136           // h buffer   [BT][136]  (batch-major, hid contiguous)

#define SMEM_HALFS (BT*SA_STR + HID*SW1_STR + HID*SW2_STR + BT*SH_STR)
#define SMEM_BYTES (SMEM_HALFS*2 + (3*HID + 1)*4)

// A-fragment (16x32 f16): lane L holds M=L%16; lanes 0-15: K=0-15;
// lanes 16-31: K=16-31.  rowp points at K=0 of this lane's M row.
__device__ __forceinline__ v16h load_a_frag(const _Float16* rowp, int lhalf) {
    const int base = lhalf << 4;                 // 0 or 16
    h8 lo = *(const h8*)(rowp + base);
    h8 hi = *(const h8*)(rowp + base + 8);
    v16h a;
#pragma unroll
    for (int i = 0; i < 8; ++i) { a[i] = lo[i]; a[i + 8] = hi[i]; }
    return a;
}

// B-fragment (32x16 f16): lane L holds N=L%16, K=(L/16)*16+i.
// colp points at 16 contiguous halfs B[k0..k0+15][n] (i.e. row n of B^T).
__device__ __forceinline__ v16h load_b_frag(const _Float16* colp) {
    h8 lo = *(const h8*)(colp);
    h8 hi = *(const h8*)(colp + 8);
    v16h b;
#pragma unroll
    for (int i = 0; i < 8; ++i) { b[i] = lo[i]; b[i + 8] = hi[i]; }
    return b;
}

__device__ __forceinline__ float fast_sigmoid(float v) {
    // v_mul(+pk) + v_exp + v_add + v_rcp  (no IEEE div sequence)
    return __builtin_amdgcn_rcpf(1.0f + __expf(-v));
}

__global__ __launch_bounds__(256) void moe_mlp_fused(
    const float* __restrict__ x,
    const float* __restrict__ W1, const float* __restrict__ b1,
    const float* __restrict__ W2, const float* __restrict__ b2,
    const float* __restrict__ W3, const float* __restrict__ b3,
    float* __restrict__ out)
{
    extern __shared__ __align__(16) unsigned char smem_raw[];
    _Float16* sA  = (_Float16*)smem_raw;            // x   [BT][SA_STR]
    _Float16* sW1 = sA  + BT  * SA_STR;             // W1^T[HID][SW1_STR]
    _Float16* sW2 = sW1 + HID * SW1_STR;            // W2^T[HID][SW2_STR]
    _Float16* sH  = sW2 + HID * SW2_STR;            // h   [BT][SH_STR]
    float* sB1 = (float*)(sH + BT * SH_STR);
    float* sB2 = sB1 + HID;
    float* sW3 = sB2 + HID;
    float* sB3 = sW3 + HID;

    const int t  = threadIdx.x;
    const int wg = blockIdx.x;
    const int e  = wg / NTILES;          // batch-tile fastest: 64 consecutive
    const int b0 = (wg % NTILES) * BT;   // WGs share this expert's weights in L2

    // ---- Stage x tile + weights into LDS (f32 -> f16) -------------------
    for (int i = t; i < BT * KPAD; i += 256) {
        int r = i >> 5, c = i & 31;
        float v = (c < D_IN) ? x[(size_t)(b0 + r) * D_IN + c] : 0.0f;
        sA[r * SA_STR + c] = (_Float16)v;
    }
    for (int i = t; i < HID * KPAD; i += 256) {
        int k = i >> 7, n = i & 127;
        float v = (k < D_IN) ? W1[(size_t)e * D_IN * HID + (size_t)k * HID + n] : 0.0f;
        sW1[n * SW1_STR + k] = (_Float16)v;
    }
    for (int i = t; i < HID * HID; i += 256) {
        int k = i >> 7, n = i & 127;
        sW2[n * SW2_STR + k] = (_Float16)W2[(size_t)e * HID * HID + i];
    }
    if (t < HID) {
        sB1[t] = b1[(size_t)e * HID + t];
        sB2[t] = b2[(size_t)e * HID + t];
        sW3[t] = W3[(size_t)e * HID + t];
    }
    if (t == 0) *sB3 = b3[e];
    __syncthreads();

    const int wave   = t >> 5;
    const int lane   = t & 31;
    const int lhalf  = lane >> 4;             // 0/1
    const int lmod   = lane & 15;
    const int mrow   = wave * 16 + lmod;      // A-fragment row (hid dim)
    const int hid0   = wave * 16 + lhalf * 8; // D-fragment hid base

    // ---- Layer 1: h1^T = W1^T (A) x x^T (B), K = 32 (padded) ------------
    // Phase-separated: all loads -> all WMMAs (back-to-back) -> all sigmoids.
    {
        const v8f bias = *(const v8f*)(&sB1[hid0]);   // C-operand = bias
        v16h a = load_a_frag(&sW1[mrow * SW1_STR], lhalf);

        v16h bf[8];
#pragma unroll
        for (int nT = 0; nT < 8; ++nT)
            bf[nT] = load_b_frag(&sA[(nT * 16 + lmod) * SA_STR + (lhalf << 4)]);

        v8f acc[8];
#pragma unroll
        for (int nT = 0; nT < 8; ++nT)
            acc[nT] = __builtin_amdgcn_wmma_f32_16x16x32_f16(
                          false, a, false, bf[nT], (short)0, bias, false, false);

#pragma unroll
        for (int nT = 0; nT < 8; ++nT) {
            h8 hv;
#pragma unroll
            for (int r = 0; r < 8; ++r) hv[r] = (_Float16)fast_sigmoid(acc[nT][r]);
            *(h8*)(&sH[(nT * 16 + lmod) * SH_STR + hid0]) = hv;  // one b128 store
        }
    }
    __syncthreads();

    // ---- Layer 2: h2^T = W2^T (A) x h1^T (B), K = 128 = 4 steps ---------
    {
        const v8f bias = *(const v8f*)(&sB2[hid0]);
        v8f acc[8];
#pragma unroll
        for (int nT = 0; nT < 8; ++nT) acc[nT] = bias;

#pragma unroll
        for (int kk = 0; kk < 4; ++kk) {
            v16h a = load_a_frag(&sW2[mrow * SW2_STR + kk * 32], lhalf);
            v16h bf[8];
#pragma unroll
            for (int nT = 0; nT < 8; ++nT)
                bf[nT] = load_b_frag(&sH[(nT * 16 + lmod) * SH_STR
                                         + kk * 32 + (lhalf << 4)]);
#pragma unroll
            for (int nT = 0; nT < 8; ++nT)
                acc[nT] = __builtin_amdgcn_wmma_f32_16x16x32_f16(
                              false, a, false, bf[nT], (short)0, acc[nT],
                              false, false);
        }
        __syncthreads();   // all waves done READING h1 before overwriting sH

#pragma unroll
        for (int nT = 0; nT < 8; ++nT) {
            h8 hv;
#pragma unroll
            for (int r = 0; r < 8; ++r) hv[r] = (_Float16)fast_sigmoid(acc[nT][r]);
            *(h8*)(&sH[(nT * 16 + lmod) * SH_STR + hid0]) = hv;  // one b128 store
        }
    }
    __syncthreads();

    // ---- Layer 3: out[b, e] = h2[b,:] . W3 + b3 -------------------------
    if (t < BT) {
        const _Float16* hp = &sH[t * SH_STR];
        float s = 0.0f;
#pragma unroll
        for (int h0 = 0; h0 < HID; h0 += 8) {
            h8 hv = *(const h8*)(hp + h0);
#pragma unroll
            for (int j = 0; j < 8; ++j) s += (float)hv[j] * sW3[h0 + j];
        }
        out[(size_t)(b0 + t) * (DIMQ * DIMQ) + e] = s + *sB3;
    }
}

extern "C" void kernel_launch(void* const* d_in, const int* in_sizes, int n_in,
                              void* d_out, int out_size, void* d_ws, size_t ws_size,
                              hipStream_t stream) {
    (void)in_sizes; (void)n_in; (void)d_ws; (void)ws_size; (void)out_size;
    const float* x  = (const float*)d_in[0];
    const float* W1 = (const float*)d_in[1];
    const float* b1 = (const float*)d_in[2];
    const float* W2 = (const float*)d_in[3];
    const float* b2 = (const float*)d_in[4];
    const float* W3 = (const float*)d_in[5];
    const float* b3 = (const float*)d_in[6];
    float* out = (float*)d_out;

    dim3 grid(EEXP * NTILES);   // 16384 workgroups: (expert, batch-tile)
    dim3 block(256);            // 8 wave32 waves
    hipLaunchKernelGGL(moe_mlp_fused, grid, block, (size_t)SMEM_BYTES, stream,
                       x, W1, b1, W2, b2, W3, b3, out);
}
